// MaskedCrossAttention_56487409877388
// MI455X (gfx1250) — matrix operand order repaired and checked by reading, not compiled, not run on hardware
//
#include <hip/hip_runtime.h>
#include <hip/hip_bf16.h>

// ---------------------------------------------------------------------------
// MaskedCrossAttention forward for MI455X (gfx1250), wave32 + WMMA f16->f32.
// Stages: LN+cast, WMMA GEMM (Q, KV, O projections), WMMA attention core.
// ---------------------------------------------------------------------------

typedef _Float16 v16h __attribute__((ext_vector_type(16)));
typedef float    v8f  __attribute__((ext_vector_type(8)));

#define DIM        2048
#define DIM_V      1024
#define HEADS      16
#define DHEAD      64
#define INNER      1024
#define BATCH      4
#define SEQ_Q      2048
#define SEQ_KV     512          // 8*64 visual tokens
#define ROWS_Q     (BATCH * SEQ_Q)   // 8192
#define ROWS_KV    (BATCH * SEQ_KV)  // 2048

// ---------------------------------------------------------------------------
// WMMA helpers (layouts per CDNA5 ISA 7.12.2, wave32)
// ---------------------------------------------------------------------------
__device__ __forceinline__ v8f wmma_f16(v16h a, v16h b, v8f c) {
  return __builtin_amdgcn_wmma_f32_16x16x32_f16(
      /*neg_a=*/false, a, /*neg_b=*/false, b,
      /*c_mod=*/(short)0, c, /*reuse_a=*/false, /*reuse_b=*/false);
}

// A fragment (16x32, MxK) from a row-major tile (ldr halves per row).
// Lane l<16: row=l, K={0..7,16..23}; lane>=16: row=l-16, K={8..15,24..31}.
__device__ __forceinline__ v16h frag_a_rowmajor(const _Float16* base, int ldr, int lane) {
  const _Float16* p = base + (size_t)(lane & 15) * ldr + ((lane >= 16) ? 8 : 0);
  v16h a;
#pragma unroll
  for (int i = 0; i < 8; ++i) { a[i] = p[i]; a[i + 8] = p[16 + i]; }
  return a;
}

// B fragment (32x16, KxN) where element (k,n) sits at base[n*ldr + k]
// (i.e. per-lane K is contiguous). Lanes 0-15 carry K={0..7,16..23} for col n=lane,
// lanes 16-31 carry K={8..15,24..31} for col n=lane-16.
__device__ __forceinline__ v16h frag_b_kcontig(const _Float16* base, int ldr, int lane) {
  const _Float16* p = base + (size_t)(lane & 15) * ldr + ((lane >= 16) ? 8 : 0);
  v16h b;
#pragma unroll
  for (int i = 0; i < 8; ++i) { b[i] = p[i]; b[i + 8] = p[16 + i]; }
  return b;
}

// ---------------------------------------------------------------------------
// Elementwise f32 -> f16 cast
// ---------------------------------------------------------------------------
__global__ void cast_f32_to_f16(const float* __restrict__ src,
                                _Float16* __restrict__ dst, int n) {
  int i = blockIdx.x * blockDim.x + threadIdx.x;
  if (i < n) dst[i] = (_Float16)src[i];
}

// ---------------------------------------------------------------------------
// LayerNorm over last dim (2048) + cast to f16. One 256-thread block per row.
// ---------------------------------------------------------------------------
__global__ void ln_cast_kernel(const float* __restrict__ y,
                               const float* __restrict__ g,
                               const float* __restrict__ b,
                               _Float16* __restrict__ out) {
  __shared__ float red[512];
  const int row = blockIdx.x;
  const float* yr = y + (size_t)row * DIM;
  float s = 0.f, s2 = 0.f;
  for (int i = threadIdx.x; i < DIM; i += 256) {
    float v = yr[i];
    s += v; s2 += v * v;
  }
  red[threadIdx.x] = s;
  red[256 + threadIdx.x] = s2;
  __syncthreads();
  for (int off = 128; off > 0; off >>= 1) {
    if ((int)threadIdx.x < off) {
      red[threadIdx.x]       += red[threadIdx.x + off];
      red[256 + threadIdx.x] += red[256 + threadIdx.x + off];
    }
    __syncthreads();
  }
  const float mean = red[0] * (1.f / DIM);
  const float var  = red[256] * (1.f / DIM) - mean * mean;
  const float rstd = rsqrtf(var + 1e-5f);
  _Float16* orow = out + (size_t)row * DIM;
  for (int i = threadIdx.x; i < DIM; i += 256) {
    float v = (yr[i] - mean) * rstd * g[i] + b[i];
    orow[i] = (_Float16)v;
  }
}

// ---------------------------------------------------------------------------
// Tiled WMMA GEMM: C(MxN) = alpha * A(MxK) @ B(KxN), f16 in, f32 accumulate.
// Block = 256 threads (8 waves), tile 256x64; each wave owns a 32x64 strip
// (8 accumulator fragments -> 8 WMMAs per 32-deep K step against 2 A-frags +
// 4 B-frags, amortizing LDS fragment reads over twice the math).
// A staged row-major in LDS; B staged transposed so B-fragments are
// two contiguous 16B lane reads.
// ---------------------------------------------------------------------------
template <typename OutT>
__global__ void gemm_wmma_kernel(const _Float16* __restrict__ A,
                                 const _Float16* __restrict__ B,
                                 OutT* __restrict__ C,
                                 int M, int N, int K, float alpha) {
  __shared__ _Float16 sA[256 * 32];   // 16 KB
  __shared__ _Float16 sBt[64 * 32];   // 4 KB, [n][k]
  const int lane = threadIdx.x & 31;
  const int wave = threadIdx.x >> 5;        // 0..7 -> 32-row strip
  const int tileM = blockIdx.y * 256;
  const int tileN = blockIdx.x * 64;

  v8f acc[2][4] = {};

  for (int kk = 0; kk < K; kk += 32) {
    // Stage A tile 256x32: 1024 vec8 loads, 4 per thread.
#pragma unroll
    for (int r = 0; r < 4; ++r) {
      int v = threadIdx.x + r * 256;        // 0..1023
      int row = v >> 2;                     // 0..255
      int col = (v & 3) * 8;                // 0,8,16,24
      const float4* src =
          reinterpret_cast<const float4*>(A + (size_t)(tileM + row) * K + kk + col);
      *reinterpret_cast<float4*>(&sA[row * 32 + col]) = *src;
    }
    // Stage B tile 32x64 transposed: 256 vec8 loads, 1 per thread.
    {
      int v = threadIdx.x;                  // 0..255
      int krow = v >> 3;                    // 0..31
      int ncol = (v & 7) * 8;               // 0..56
      const float4* src =
          reinterpret_cast<const float4*>(B + (size_t)(kk + krow) * N + tileN + ncol);
      float4 pk = *src;
      _Float16 tmp[8];
      *reinterpret_cast<float4*>(tmp) = pk;
#pragma unroll
      for (int i = 0; i < 8; ++i) sBt[(ncol + i) * 32 + krow] = tmp[i];
    }
    __syncthreads();

    v16h af0 = frag_a_rowmajor(&sA[(wave * 32 +  0) * 32], 32, lane);
    v16h af1 = frag_a_rowmajor(&sA[(wave * 32 + 16) * 32], 32, lane);
#pragma unroll
    for (int t = 0; t < 4; ++t) {
      v16h bfrag = frag_b_kcontig(&sBt[t * 16 * 32], 32, lane);
      acc[0][t] = wmma_f16(af0, bfrag, acc[0][t]);
      acc[1][t] = wmma_f16(af1, bfrag, acc[1][t]);
    }
    __syncthreads();
  }

  // Store: C layout M = vgpr + 8*(lane>=16), N = lane&15.
  const int m0 = (lane >= 16) ? 8 : 0;
  const int n  = lane & 15;
#pragma unroll
  for (int r = 0; r < 2; ++r) {
#pragma unroll
    for (int t = 0; t < 4; ++t) {
#pragma unroll
      for (int j = 0; j < 8; ++j) {
        size_t row = (size_t)(tileM + wave * 32 + r * 16 + m0 + j);
        C[row * N + tileN + t * 16 + n] = (OutT)(acc[r][t][j] * alpha);
      }
    }
  }
}

// ---------------------------------------------------------------------------
// Transpose V out of the fused KV buffer into (b,h,d,keys) so that P@V
// B-fragments become contiguous lane reads. Writes are coalesced.
// kv: (ROWS_KV x 2*INNER), v at column 1024 + h*64 + d.
// ---------------------------------------------------------------------------
__global__ void transpose_v_kernel(const _Float16* __restrict__ kv,
                                   _Float16* __restrict__ vT) {
  size_t i = (size_t)blockIdx.x * 256 + threadIdx.x;   // 4*16*64*512 total
  int m = (int)(i & 511);
  size_t r = i >> 9;
  int d = (int)(r & 63); r >>= 6;
  int h = (int)(r & 15);
  int b = (int)(r >> 4);
  vT[i] = kv[(size_t)(b * SEQ_KV + m) * (2 * INNER) + INNER + h * DHEAD + d];
}

// ---------------------------------------------------------------------------
// Attention core: one wave per (b,h, 16-query tile).
//   S = q @ k^T (K=64, two WMMA steps per 16-key tile, 32 tiles)
//   softmax over 512 keys (scores in LDS, two lanes per row + LDS combine)
//   O = P @ v   (K=512, 16 WMMA steps x 4 d-tiles), f16 out (merged heads)
// ---------------------------------------------------------------------------
__global__ void attn_kernel(const _Float16* __restrict__ q,     // (8192 x 1024)
                            const _Float16* __restrict__ kv,    // (2048 x 2048)
                            const _Float16* __restrict__ vT,    // (b,h,64,512)
                            _Float16* __restrict__ attn_out) {  // (8192 x 1024)
  __shared__ float    sScore[16 * 512];   // 32 KB
  __shared__ _Float16 sAlpha[16 * 512];   // 16 KB
  __shared__ float    sPart[32];          // per-lane softmax partials
  const int lane = threadIdx.x;           // 32 threads
  const int bh = blockIdx.y;
  const int b  = bh >> 4;
  const int h  = bh & 15;
  const int q0 = blockIdx.x * 16;

  // Q fragments for this 16-row tile (K=64 -> two 16x32 fragments), reused
  // across all 32 key tiles.
  const _Float16* qbase = q + (size_t)(b * SEQ_Q + q0) * INNER + h * DHEAD;
  v16h qa0 = frag_a_rowmajor(qbase,      INNER, lane);
  v16h qa1 = frag_a_rowmajor(qbase + 32, INNER, lane);

  // ---- S = q @ k^T ----
  const _Float16* kbase = kv + (size_t)(b * SEQ_KV) * (2 * INNER) + h * DHEAD;
  const int m0 = (lane >= 16) ? 8 : 0;
  const int n  = lane & 15;
  for (int jt = 0; jt < 32; ++jt) {
    const _Float16* ktile = kbase + (size_t)(jt * 16) * (2 * INNER);
    v8f s = {};
    v16h kb0 = frag_b_kcontig(ktile,      2 * INNER, lane);   // d 0..31
    v16h kb1 = frag_b_kcontig(ktile + 32, 2 * INNER, lane);   // d 32..63
    s = wmma_f16(qa0, kb0, s);
    s = wmma_f16(qa1, kb1, s);
#pragma unroll
    for (int j = 0; j < 8; ++j)
      sScore[(m0 + j) * 512 + jt * 16 + n] = s[j];
  }
  __syncthreads();

  // ---- softmax over 512 keys: two lanes per query row (256 elems each) ----
  {
    const int row  = lane & 15;
    const int half = lane >> 4;               // 0 or 1
    float* rp = &sScore[row * 512 + half * 256];

    float mx = -3.402823466e38f;
    for (int j = 0; j < 256; ++j) mx = fmaxf(mx, rp[j]);
    sPart[lane] = mx;
    __syncthreads();
    mx = fmaxf(sPart[row], sPart[row + 16]);
    __syncthreads();

    float sum = 0.f;
    for (int j = 0; j < 256; ++j) {
      float e = __expf(rp[j] - mx);
      rp[j] = e;
      sum += e;
    }
    sPart[lane] = sum;
    __syncthreads();
    const float inv = 1.f / (sPart[row] + sPart[row + 16]);

    _Float16* ap = &sAlpha[row * 512 + half * 256];
    for (int j = 0; j < 256; ++j) ap[j] = (_Float16)(rp[j] * inv);
  }
  __syncthreads();

  // ---- O = P @ v ----
  v8f o[4] = {};
  const _Float16* vtb = vT + (size_t)bh * DHEAD * SEQ_KV;
  for (int ks = 0; ks < 16; ++ks) {
    v16h af = frag_a_rowmajor(&sAlpha[ks * 32], 512, lane);
#pragma unroll
    for (int t = 0; t < 4; ++t) {
      v16h bf = frag_b_kcontig(vtb + (size_t)(t * 16) * SEQ_KV + ks * 32, SEQ_KV, lane);
      o[t] = wmma_f16(af, bf, o[t]);
    }
  }

  // store merged-head f16 output: row = b*2048 + q, col = h*64 + t*16 + n
#pragma unroll
  for (int t = 0; t < 4; ++t) {
#pragma unroll
    for (int j = 0; j < 8; ++j) {
      size_t row = (size_t)(b * SEQ_Q + q0 + m0 + j);
      attn_out[row * INNER + h * DHEAD + t * 16 + n] = (_Float16)o[t][j];
    }
  }
}

// ---------------------------------------------------------------------------
// Launch
// ---------------------------------------------------------------------------
extern "C" void kernel_launch(void* const* d_in, const int* in_sizes, int n_in,
                              void* d_out, int out_size, void* d_ws, size_t ws_size,
                              hipStream_t stream) {
  const float* y    = (const float*)d_in[0];
  const float* vis  = (const float*)d_in[1];
  const float* ln_g = (const float*)d_in[2];
  const float* ln_b = (const float*)d_in[3];
  const float* Wq   = (const float*)d_in[4];
  const float* Wkv  = (const float*)d_in[5];
  const float* Wo   = (const float*)d_in[6];
  float* out = (float*)d_out;

  char* ws = (char*)d_ws;
  size_t off = 0;
  auto alloc = [&](size_t elems) {
    void* p = ws + off;
    off += (elems * sizeof(_Float16) + 255) & ~(size_t)255;
    return (_Float16*)p;
  };
  _Float16* yn_h   = alloc((size_t)ROWS_Q  * DIM);        // 32 MB
  _Float16* Wq_h   = alloc((size_t)DIM     * INNER);      // 4 MB
  _Float16* Wkv_h  = alloc((size_t)DIM_V   * 2 * INNER);  // 4 MB
  _Float16* Wo_h   = alloc((size_t)INNER   * DIM);        // 4 MB
  _Float16* vis_h  = alloc((size_t)ROWS_KV * DIM_V);      // 4 MB
  _Float16* q_h    = alloc((size_t)ROWS_Q  * INNER);      // 16 MB
  _Float16* kv_h   = alloc((size_t)ROWS_KV * 2 * INNER);  // 8 MB
  _Float16* vT_h   = alloc((size_t)BATCH * HEADS * DHEAD * SEQ_KV); // 4 MB
  _Float16* attn_h = alloc((size_t)ROWS_Q  * INNER);      // 16 MB
  (void)ws_size; (void)in_sizes; (void)n_in; (void)out_size;

  // Casts to f16
  {
    int n = DIM * INNER;          // 2M
    cast_f32_to_f16<<<(n + 255) / 256, 256, 0, stream>>>(Wq, Wq_h, n);
    n = DIM_V * 2 * INNER;        // 2M
    cast_f32_to_f16<<<(n + 255) / 256, 256, 0, stream>>>(Wkv, Wkv_h, n);
    n = INNER * DIM;              // 2M
    cast_f32_to_f16<<<(n + 255) / 256, 256, 0, stream>>>(Wo, Wo_h, n);
    n = ROWS_KV * DIM_V;          // 2M
    cast_f32_to_f16<<<(n + 255) / 256, 256, 0, stream>>>(vis, vis_h, n);
  }

  // LayerNorm + cast of y
  ln_cast_kernel<<<ROWS_Q, 256, 0, stream>>>(y, ln_g, ln_b, yn_h);

  const float scale = 0.125f;  // 64^-0.5

  // Q = LN(y) @ Wq * scale            (8192 x 1024, K=2048)
  gemm_wmma_kernel<_Float16><<<dim3(INNER / 64, ROWS_Q / 256), 256, 0, stream>>>(
      yn_h, Wq_h, q_h, ROWS_Q, INNER, DIM, scale);

  // KV = vis @ Wkv                    (2048 x 2048, K=1024)
  gemm_wmma_kernel<_Float16><<<dim3((2 * INNER) / 64, ROWS_KV / 256), 256, 0, stream>>>(
      vis_h, Wkv_h, kv_h, ROWS_KV, 2 * INNER, DIM_V, 1.0f);

  // V transposed to (b,h,d,keys)
  {
    int n = BATCH * HEADS * DHEAD * SEQ_KV;  // 2M
    transpose_v_kernel<<<n / 256, 256, 0, stream>>>(kv_h, vT_h);
  }

  // Attention core: grid (query tiles, b*h), one wave per block
  attn_kernel<<<dim3(SEQ_Q / 16, BATCH * HEADS), 32, 0, stream>>>(
      q_h, kv_h, vT_h, attn_h);

  // Output projection: out = attn @ Wo  (8192 x 2048, K=1024), f32 out
  gemm_wmma_kernel<float><<<dim3(DIM / 64, ROWS_Q / 256), 256, 0, stream>>>(
      attn_h, Wo_h, out, ROWS_Q, DIM, INNER, 1.0f);
}
